// MultiHeadDiffAttention_14654428414299
// MI455X (gfx1250) — matrix operand order repaired and checked by reading, not compiled, not run on hardware
//
#include <hip/hip_runtime.h>

typedef unsigned int   u32;
typedef unsigned short u16;
typedef __attribute__((ext_vector_type(16))) __bf16 v16bf;
typedef __attribute__((ext_vector_type(8)))  float  v8f;

union AFrag { v16bf v; u32 u[8]; };

// ---------------------------------------------------------------- constants
// LAYER_IDX=3: LAM_INIT = 0.8 - 0.6*exp(-0.6)
#define LAM_INIT_F      0.47071301834358414f
#define ONE_MINUS_LAM_F 0.52928698165641586f
#define NBATCH 4
#define SEQ    2048
#define CEMB   384
#define NHEAD  6
#define HDIM   64
#define VDIM   128     // 2*D
#define FDIM   768     // 2*C
// score scale folded with log2(e): (1/sqrt(64)) * 1.4426950408889634
#define SCALE_LOG2E     0.18033688011112043f

// ---------------------------------------------------------------- helpers
__device__ __forceinline__ v8f wmma_bf16(v16bf a, v16bf b, v8f c) {
  return __builtin_amdgcn_wmma_f32_16x16x32_bf16(false, a, false, b,
                                                 (short)0, c, false, false);
}

__device__ __forceinline__ float fexp2(float x) {
  return __builtin_amdgcn_exp2f(x);   // single v_exp_f32 (base-2 native)
}

__device__ __forceinline__ u16 f2bf(float f) {
  u32 u = __float_as_uint(f);
  u32 r = u + 0x7FFFu + ((u >> 16) & 1u);   // round-to-nearest-even
  return (u16)(r >> 16);
}

__device__ __forceinline__ float redmax16(float x) {
  x = fmaxf(x, __shfl_xor(x, 1, 32));
  x = fmaxf(x, __shfl_xor(x, 2, 32));
  x = fmaxf(x, __shfl_xor(x, 4, 32));
  x = fmaxf(x, __shfl_xor(x, 8, 32));
  return x;
}
__device__ __forceinline__ float redsum16(float x) {
  x += __shfl_xor(x, 1, 32);
  x += __shfl_xor(x, 2, 32);
  x += __shfl_xor(x, 4, 32);
  x += __shfl_xor(x, 8, 32);
  return x;
}

// A-fragment (16x32 bf16) per-VGPR K start for this lane-half
__device__ __forceinline__ int a_k0(int v, int hh) {
  return hh * 16 + 2 * v;
}

// ---------------------------------------------------------------- fp32->bf16
__global__ void cvt_f32_bf16(const float* __restrict__ src,
                             u16* __restrict__ dst, int n) {
  int i = blockIdx.x * blockDim.x + threadIdx.x;
  if (i < n) dst[i] = f2bf(src[i]);
}

// ---------------------------------------------------------------- lambda[h]
__global__ __launch_bounds__(32)
void lam_kernel(const float* __restrict__ lq1, const float* __restrict__ lk1,
                const float* __restrict__ lq2, const float* __restrict__ lk2,
                float* __restrict__ lam) {
  int lane = threadIdx.x;
  for (int h = 0; h < NHEAD; ++h) {
    int i0 = h * 64 + lane, i1 = i0 + 32;
    float s1 = lq1[i0] * lk1[i0] + lq1[i1] * lk1[i1];
    float s2 = lq2[i0] * lk2[i0] + lq2[i1] * lk2[i1];
    #pragma unroll
    for (int m = 1; m <= 16; m <<= 1) {
      s1 += __shfl_xor(s1, m, 32);
      s2 += __shfl_xor(s2, m, 32);
    }
    if (lane == 0) lam[h] = __expf(s1) - __expf(s2) + LAM_INIT_F;
  }
}

// ---------------------------------------------------------------- projections
// out(8192 x 2304) = xbf(8192 x 384) @ Wall(2304 x 384)^T, routed to
// Q1/Q2/K1/K2 (B,H,T,64) and Vt (B,H,128,T), all bf16.
__global__ __launch_bounds__(32)
void proj_kernel(const u16* __restrict__ xbf, const u16* __restrict__ Wall,
                 u16* __restrict__ Q1, u16* __restrict__ Q2,
                 u16* __restrict__ K1, u16* __restrict__ K2,
                 u16* __restrict__ Vt) {
  int rt = blockIdx.x;            // 512 row tiles of 16
  int ot = blockIdx.y;            // 144 output tiles of 16
  int lane = threadIdx.x;
  int n  = lane & 15;
  int hh = lane >> 4;

  const u16* ap  = xbf + (size_t)(rt * 16 + n) * CEMB;
  const u16* bp0 = Wall + (size_t)(ot * 16 + n) * CEMB + hh * 16;

  v8f acc = {0.f, 0.f, 0.f, 0.f, 0.f, 0.f, 0.f, 0.f};
  #pragma unroll
  for (int kc = 0; kc < CEMB / 32; ++kc) {
    AFrag a;
    #pragma unroll
    for (int v = 0; v < 8; ++v)
      a.u[v] = *(const u32*)(ap + kc * 32 + a_k0(v, hh));
    v16bf bw = *(const v16bf*)(bp0 + kc * 32);
    acc = wmma_bf16(a.v, bw, acc);
  }

  int o = ot * 16 + n;            // output feature (uniform segment per tile)
  if (o < 4 * CEMB) {             // Q1/Q2/K1/K2
    u16* dst = (o < CEMB) ? Q1 : (o < 2 * CEMB) ? Q2
             : (o < 3 * CEMB) ? K1 : K2;
    int oo = o % CEMB;
    int h = oo >> 6, d = oo & 63;
    #pragma unroll
    for (int v = 0; v < 8; ++v) {
      int row = rt * 16 + v + hh * 8;
      int b = row >> 11, t = row & 2047;
      dst[((size_t)(b * NHEAD + h) * SEQ + t) * HDIM + d] = f2bf(acc[v]);
    }
  } else {                        // V -> transposed layout
    int f = o - 4 * CEMB;
    int h = f >> 7, vd = f & 127;
    #pragma unroll
    for (int v = 0; v < 8; ++v) {
      int row = rt * 16 + v + hh * 8;
      int b = row >> 11, t = row & 2047;
      Vt[((size_t)(b * NHEAD + h) * VDIM + vd) * SEQ + t] = f2bf(acc[v]);
    }
  }
}

// ---------------------------------------------------------------- attn chunk
// One 32-key chunk of the dual flash-softmax loop. MASKED only for the
// single diagonal chunk (all earlier chunks are fully inside the causal
// triangle for every row of the 16-query tile).
template<bool MASKED>
__device__ __forceinline__
void attn_chunk(int kb, int qbase, int n, int hh, int bh,
                const u16* __restrict__ K1, const u16* __restrict__ K2,
                const u16* __restrict__ Vt,
                const AFrag (&aq1)[2], const AFrag (&aq2)[2],
                v8f (&acc1)[8], v8f (&acc2)[8],
                float (&M1)[8], float (&L1)[8],
                float (&M2)[8], float (&L2)[8],
                u16 (*Pl1)[32], u16 (*Pl2)[32]) {
  // ---- scores: two 16x16 tiles (key groups), K-dim 64 = 2 WMMAs each
  v8f s1[2], s2[2];
  #pragma unroll
  for (int g = 0; g < 2; ++g) {
    int key = kb + g * 16 + n;
    const u16* k1p = K1 + ((size_t)bh * SEQ + key) * HDIM + hh * 16;
    const u16* k2p = K2 + ((size_t)bh * SEQ + key) * HDIM + hh * 16;
    v8f c1 = {0.f, 0.f, 0.f, 0.f, 0.f, 0.f, 0.f, 0.f};
    v8f c2 = c1;
    c1 = wmma_bf16(aq1[0].v, *(const v16bf*)(k1p), c1);
    c1 = wmma_bf16(aq1[1].v, *(const v16bf*)(k1p + 32), c1);
    c2 = wmma_bf16(aq2[0].v, *(const v16bf*)(k2p), c2);
    c2 = wmma_bf16(aq2[1].v, *(const v16bf*)(k2p + 32), c2);
    s1[g] = c1; s2[g] = c2;
  }

  // ---- dual online softmax in exp2 domain (row stats: 16 lanes per VGPR)
  #pragma unroll
  for (int v = 0; v < 8; ++v) {
    float a0 = s1[0][v] * SCALE_LOG2E;
    float a1 = s1[1][v] * SCALE_LOG2E;
    float b0 = s2[0][v] * SCALE_LOG2E;
    float b1 = s2[1][v] * SCALE_LOG2E;
    if (MASKED) {
      int q = qbase + v + hh * 8;
      bool v0 = (kb + n) <= q, v1 = (kb + 16 + n) <= q;
      a0 = v0 ? a0 : -3.0e38f;
      a1 = v1 ? a1 : -3.0e38f;
      b0 = v0 ? b0 : -3.0e38f;
      b1 = v1 ? b1 : -3.0e38f;
    }

    float Mn1 = fmaxf(M1[v], redmax16(fmaxf(a0, a1)));
    float al1 = fexp2(M1[v] - Mn1);
    float p0 = fexp2(a0 - Mn1), p1 = fexp2(a1 - Mn1);
    L1[v] = L1[v] * al1 + redsum16(p0 + p1);
    M1[v] = Mn1;
    Pl1[v + hh * 8][n]      = f2bf(p0);
    Pl1[v + hh * 8][16 + n] = f2bf(p1);

    float Mn2 = fmaxf(M2[v], redmax16(fmaxf(b0, b1)));
    float al2 = fexp2(M2[v] - Mn2);
    float r0 = fexp2(b0 - Mn2), r1 = fexp2(b1 - Mn2);
    L2[v] = L2[v] * al2 + redsum16(r0 + r1);
    M2[v] = Mn2;
    Pl2[v + hh * 8][n]      = f2bf(r0);
    Pl2[v + hh * 8][16 + n] = f2bf(r1);

    #pragma unroll
    for (int nt = 0; nt < 8; ++nt) {
      acc1[nt][v] *= al1;
      acc2[nt][v] *= al2;
    }
  }
  __syncthreads();                // LDS transpose bounce (C-layout -> A-layout)

  AFrag pa1, pa2;
  #pragma unroll
  for (int v = 0; v < 8; ++v) {
    int k0 = a_k0(v, hh);
    pa1.u[v] = *(const u32*)&Pl1[n][k0];
    pa2.u[v] = *(const u32*)&Pl2[n][k0];
  }

  // ---- P @ V over 8 output tiles of 16 (Vt rows are contiguous in keys)
  const u16* vp0 = Vt + ((size_t)bh * VDIM + n) * SEQ + kb + hh * 16;
  #pragma unroll
  for (int nt = 0; nt < 8; ++nt) {
    v16bf bv = *(const v16bf*)(vp0 + (size_t)nt * 16 * SEQ);
    acc1[nt] = wmma_bf16(pa1.v, bv, acc1[nt]);
    acc2[nt] = wmma_bf16(pa2.v, bv, acc2[nt]);
  }
  __syncthreads();                // protect Pl reuse next chunk
}

// ---------------------------------------------------------------- attention
// One wave per (b,h,16-query tile). Dual flash softmax + fused LayerNorm.
__global__ __launch_bounds__(32)
void attn_kernel(const u16* __restrict__ Q1, const u16* __restrict__ Q2,
                 const u16* __restrict__ K1, const u16* __restrict__ K2,
                 const u16* __restrict__ Vt, const float* __restrict__ lam,
                 u16* __restrict__ Y) {
  int qt = blockIdx.x;            // 128 query tiles
  int h  = blockIdx.y;            // 6 heads
  int b  = blockIdx.z;            // 4 batches
  int bh = b * NHEAD + h;
  int lane = threadIdx.x;
  int n  = lane & 15;
  int hh = lane >> 4;
  int qbase = qt * 16;
  float lam_h = lam[h];

  __shared__ u16 Pl1[16][32];
  __shared__ u16 Pl2[16][32];

  // Q A-fragments: 2 dim-chunks of 32 for each of Q1,Q2
  AFrag aq1[2], aq2[2];
  {
    const u16* q1p = Q1 + ((size_t)bh * SEQ + qbase + n) * HDIM;
    const u16* q2p = Q2 + ((size_t)bh * SEQ + qbase + n) * HDIM;
    #pragma unroll
    for (int f = 0; f < 2; ++f)
      #pragma unroll
      for (int v = 0; v < 8; ++v) {
        int k0 = 32 * f + a_k0(v, hh);
        aq1[f].u[v] = *(const u32*)(q1p + k0);
        aq2[f].u[v] = *(const u32*)(q2p + k0);
      }
  }

  v8f acc1[8], acc2[8];
  float M1[8], L1[8], M2[8], L2[8];
  #pragma unroll
  for (int nt = 0; nt < 8; ++nt) {
    v8f z = {0.f, 0.f, 0.f, 0.f, 0.f, 0.f, 0.f, 0.f};
    acc1[nt] = z; acc2[nt] = z;
  }
  #pragma unroll
  for (int v = 0; v < 8; ++v) {
    M1[v] = -3.0e38f; M2[v] = -3.0e38f; L1[v] = 0.f; L2[v] = 0.f;
  }

  // 32-key chunks covering keys <= qbase+15; only the last chunk crosses
  // the causal diagonal, all earlier chunks are fully valid.
  int nch = (qt + 2) >> 1;
  for (int kc = 0; kc < nch - 1; ++kc)
    attn_chunk<false>(kc * 32, qbase, n, hh, bh, K1, K2, Vt, aq1, aq2,
                      acc1, acc2, M1, L1, M2, L2, Pl1, Pl2);
  attn_chunk<true>((nch - 1) * 32, qbase, n, hh, bh, K1, K2, Vt, aq1, aq2,
                   acc1, acc2, M1, L1, M2, L2, Pl1, Pl2);

  // ---- combine, per-head LayerNorm over 128, store bf16 Y (B,T,768)
  float y[8][8];                  // [nt][v]
  #pragma unroll
  for (int v = 0; v < 8; ++v) {
    float inv1 = 1.0f / L1[v];
    float inv2 = lam_h / L2[v];
    #pragma unroll
    for (int nt = 0; nt < 8; ++nt)
      y[nt][v] = acc1[nt][v] * inv1 - acc2[nt][v] * inv2;
  }
  #pragma unroll
  for (int v = 0; v < 8; ++v) {
    float s = 0.f, ss = 0.f;
    #pragma unroll
    for (int nt = 0; nt < 8; ++nt) { s += y[nt][v]; ss += y[nt][v] * y[nt][v]; }
    s = redsum16(s); ss = redsum16(ss);
    float mu = s * (1.0f / 128.0f);
    float var = ss * (1.0f / 128.0f) - mu * mu;
    float rs = rsqrtf(var + 1e-5f) * ONE_MINUS_LAM_F;
    int q = qbase + v + hh * 8;
    size_t base = ((size_t)(b * SEQ + q)) * FDIM + h * VDIM + n;
    #pragma unroll
    for (int nt = 0; nt < 8; ++nt)
      Y[base + nt * 16] = f2bf((y[nt][v] - mu) * rs);
  }
}

// ---------------------------------------------------------------- out proj
// out(8192 x 384) f32 = Y(8192 x 768) bf16 @ Wc(384 x 768)^T bf16
__global__ __launch_bounds__(32)
void outproj_kernel(const u16* __restrict__ Y, const u16* __restrict__ Wcb,
                    float* __restrict__ out) {
  int rt = blockIdx.x;            // 512 row tiles
  int ot = blockIdx.y;            // 24 output tiles
  int lane = threadIdx.x;
  int n  = lane & 15;
  int hh = lane >> 4;

  const u16* ap  = Y + (size_t)(rt * 16 + n) * FDIM;
  const u16* bp0 = Wcb + (size_t)(ot * 16 + n) * FDIM + hh * 16;

  v8f acc = {0.f, 0.f, 0.f, 0.f, 0.f, 0.f, 0.f, 0.f};
  #pragma unroll
  for (int kc = 0; kc < FDIM / 32; ++kc) {
    AFrag a;
    #pragma unroll
    for (int v = 0; v < 8; ++v)
      a.u[v] = *(const u32*)(ap + kc * 32 + a_k0(v, hh));
    v16bf bw = *(const v16bf*)(bp0 + kc * 32);
    acc = wmma_bf16(a.v, bw, acc);
  }
  #pragma unroll
  for (int v = 0; v < 8; ++v)
    out[(size_t)(rt * 16 + v + hh * 8) * CEMB + ot * 16 + n] = acc[v];
}

// ---------------------------------------------------------------- launch
extern "C" void kernel_launch(void* const* d_in, const int* in_sizes, int n_in,
                              void* d_out, int out_size, void* d_ws, size_t ws_size,
                              hipStream_t stream) {
  const float* x   = (const float*)d_in[0];
  const float* Wq1 = (const float*)d_in[1];
  const float* Wq2 = (const float*)d_in[2];
  const float* Wk1 = (const float*)d_in[3];
  const float* Wk2 = (const float*)d_in[4];
  const float* Wv  = (const float*)d_in[5];
  const float* Wc  = (const float*)d_in[6];
  const float* lq1 = (const float*)d_in[7];
  const float* lk1 = (const float*)d_in[8];
  const float* lq2 = (const float*)d_in[9];
  const float* lk2 = (const float*)d_in[10];
  float* out = (float*)d_out;

  // workspace carve-up (all bf16 buffers, 32B-aligned offsets)
  u16* xbf  = (u16*)d_ws;                 // 8192*384
  u16* Wall = xbf + 8192 * 384;           // 2304*384
  u16* Wcb  = Wall + 2304 * 384;          // 384*768
  u16* Q1   = Wcb + 384 * 768;            // 4*6*2048*64 each
  u16* Q2   = Q1 + 3145728;
  u16* K1   = Q2 + 3145728;
  u16* K2   = K1 + 3145728;
  u16* Vt   = K2 + 3145728;               // 4*6*128*2048
  u16* Ybf  = Vt + 6291456;               // 8192*768
  float* lam = (float*)(Ybf + 6291456);   // 6 floats

  // fp32 -> bf16 conversions
  cvt_f32_bf16<<<(8192 * 384 + 255) / 256, 256, 0, stream>>>(x, xbf, 8192 * 384);
  cvt_f32_bf16<<<(147456 + 255) / 256, 256, 0, stream>>>(Wq1, Wall + 0,      147456);
  cvt_f32_bf16<<<(147456 + 255) / 256, 256, 0, stream>>>(Wq2, Wall + 147456, 147456);
  cvt_f32_bf16<<<(147456 + 255) / 256, 256, 0, stream>>>(Wk1, Wall + 294912, 147456);
  cvt_f32_bf16<<<(147456 + 255) / 256, 256, 0, stream>>>(Wk2, Wall + 442368, 147456);
  cvt_f32_bf16<<<(294912 + 255) / 256, 256, 0, stream>>>(Wv,  Wall + 589824, 294912);
  cvt_f32_bf16<<<(294912 + 255) / 256, 256, 0, stream>>>(Wc,  Wcb,           294912);

  lam_kernel<<<1, 32, 0, stream>>>(lq1, lk1, lq2, lk2, lam);

  proj_kernel<<<dim3(512, 144), 32, 0, stream>>>(xbf, Wall, Q1, Q2, K1, K2, Vt);

  attn_kernel<<<dim3(128, 6, 4), 32, 0, stream>>>(Q1, Q2, K1, K2, Vt, lam, Ybf);

  outproj_kernel<<<dim3(512, 24), 32, 0, stream>>>(Ybf, Wcb, out);

  (void)in_sizes; (void)n_in; (void)out_size; (void)ws_size;
}
